// PlayerEncoder_80315888435663
// MI455X (gfx1250) — compile-verified, matrix operand order checked by reading, not run on hardware
//
#include <hip/hip_runtime.h>
#include <hip/hip_bf16.h>

// ---------------------------------------------------------------------------
// PlayerEncoder for MI455X (gfx1250): GAT x2 + projections.
// Matrix math in f16 via V_WMMA_F32_16X16X32_F16 (f32 accumulate).
// ---------------------------------------------------------------------------

typedef __attribute__((ext_vector_type(16))) _Float16 v16h;
typedef __attribute__((ext_vector_type(8)))  _Float16 v8h;
typedef __attribute__((ext_vector_type(8)))  float    v8f;

#define BATCH 8
#define NB    1024          // agents per batch row
#define HID   256
#define HEADS 4
#define DH    64
#define ED    31            // entity_dim
#define MD    35            // merged_dim
#define ROWS  (BATCH * NB)  // 8192
#define OUT_EMB_OFF ((size_t)ROWS * HID)   // my_emb offset in d_out (floats)

// ---------------------------------------------------------------------------
// K1: merged features -> proj = merged @ Wp + bp  (K=35, scalar), plus mask.
// grid ROWS blocks x 256 threads.
// ---------------------------------------------------------------------------
__global__ void prep_proj(const float* __restrict__ agents,
                          const float* __restrict__ Wp,
                          const float* __restrict__ bp,
                          float* __restrict__ projF32,
                          _Float16* __restrict__ projF16,
                          float* __restrict__ mask) {
  int row = blockIdx.x;          // b*NB + n
  int t   = threadIdx.x;         // 0..255 output column
  __shared__ float sm[MD];
  const float* ag = agents + (size_t)row * ED;
  if (t == 0) sm[0] = ag[0];                              // id
  if (t >= 1 && t < 6) {                                  // one-hot npc type
    int npc = (int)ag[1];
    sm[t] = (npc == (t - 1)) ? 1.f : 0.f;
  }
  if (t >= 6 && t < MD) sm[t] = ag[2 + (t - 6)];          // remaining feats
  __syncthreads();
  float acc = bp[t];
  for (int k = 0; k < MD; ++k) acc += sm[k] * Wp[k * HID + t];
  projF32[(size_t)row * HID + t] = acc;
  projF16[(size_t)row * HID + t] = (_Float16)acc;
  if (t == 0) mask[row] = (ag[0] != 0.f) ? 1.f : 0.f;
}

// ---------------------------------------------------------------------------
// K2: convert weights to f16, transposed (Bt[n][k] = W[k][n]) for WMMA B path.
// ---------------------------------------------------------------------------
__global__ void conv_weights(const float* __restrict__ W1,
                             const float* __restrict__ W2,
                             const float* __restrict__ Wa,
                             _Float16* __restrict__ W1t,
                             _Float16* __restrict__ W2t,
                             _Float16* __restrict__ Wat) {
  int idx = blockIdx.x * blockDim.x + threadIdx.x;
  if (idx < 65536) {
    int n = idx >> 8, k = idx & 255;
    W1t[n * 256 + k] = (_Float16)W1[k * 256 + n];
  } else if (idx < 131072) {
    int e = idx - 65536; int n = e >> 8, k = e & 255;
    W2t[n * 256 + k] = (_Float16)W2[k * 256 + n];
  } else if (idx < 131072 + 16384) {
    int e = idx - 131072; int n = e >> 6, k = e & 63;
    Wat[n * 64 + k] = (_Float16)Wa[k * 256 + n];
  }
}

// ---------------------------------------------------------------------------
// K3: WMMA GEMM.  C[M,N] = A[M,K](f16,row-major) x Bt[N,K](f16,= B^T).
// One wave computes one 16x16 tile, K in chunks of 32.
// A-operand lane layout (ISA 7.12.2, 16-bit A 16x32): lane l -> row (l&15),
// halves [0..7] = K(koff..koff+7), [8..15] = K(16+koff..), koff=(l>>4)*8.
// B^T rows load with the identical pattern (n plays the role of m).
// mode 0: out f32 = acc + bias   (agent_embeddings)
// mode 1: out f16 row-major (hN) AND per-head transposed f16 hT[b,h,d,j]
// ---------------------------------------------------------------------------
__global__ void wmma_gemm16(const _Float16* __restrict__ A,
                            const _Float16* __restrict__ Bt,
                            const float* __restrict__ bias,
                            float* __restrict__ outF32,
                            _Float16* __restrict__ outF16,
                            _Float16* __restrict__ outT,
                            int M, int N, int K, int mode) {
  int wave = blockIdx.x * (blockDim.x >> 5) + (threadIdx.x >> 5);
  int lane = threadIdx.x & 31;
  int tilesN = N >> 4;
  int tm = wave / tilesN;
  int tn = wave - tm * tilesN;
  if (tm * 16 >= M) return;
  int l15  = lane & 15;
  int koff = (lane >> 4) * 8;
  const _Float16* arow = A  + (size_t)(tm * 16 + l15) * K;
  const _Float16* brow = Bt + (size_t)(tn * 16 + l15) * K;
  v8f acc = {};
  for (int k0 = 0; k0 < K; k0 += 32) {
    v8h alo = *(const v8h*)(arow + k0 + koff);
    v8h ahi = *(const v8h*)(arow + k0 + 16 + koff);
    v8h blo = *(const v8h*)(brow + k0 + koff);
    v8h bhi = *(const v8h*)(brow + k0 + 16 + koff);
    v16h a, b;
#pragma unroll
    for (int q = 0; q < 8; ++q) {
      a[q] = alo[q]; a[8 + q] = ahi[q];
      b[q] = blo[q]; b[8 + q] = bhi[q];
    }
    acc = __builtin_amdgcn_wmma_f32_16x16x32_f16(false, a, false, b,
                                                 (short)0, acc, false, false);
  }
  // D layout: lane l holds column n=(l&15); VGPR v holds row (l>>4)*8 + v.
  int mbase = (lane >> 4) * 8;
  int ng = tn * 16 + l15;
#pragma unroll
  for (int v = 0; v < 8; ++v) {
    int mg = tm * 16 + mbase + v;
    float val = acc[v];
    if (mode == 0) {
      outF32[(size_t)mg * N + ng] = val + (bias ? bias[ng] : 0.f);
    } else {
      outF16[(size_t)mg * N + ng] = (_Float16)val;
      int bb = mg >> 10, j = mg & (NB - 1);
      int hh = ng >> 6,  d = ng & 63;
      outT[(((size_t)(bb * HEADS + hh) * DH + d) << 10) + j] = (_Float16)val;
    }
  }
}

// ---------------------------------------------------------------------------
// K4: attention scalar scores e_i = <h, a_src>, e_j = <h, a_dst>.
// one thread per (b,n,head).
// ---------------------------------------------------------------------------
__global__ void compute_e(const _Float16* __restrict__ hN,
                          const float* __restrict__ a_src,
                          const float* __restrict__ a_dst,
                          float* __restrict__ e_i, float* __restrict__ e_j,
                          int total) {
  int idx = blockIdx.x * blockDim.x + threadIdx.x;
  if (idx >= total) return;
  int h  = idx & (HEADS - 1);
  int bn = idx >> 2;
  const _Float16* row = hN + (size_t)bn * HID + h * DH;
  float si = 0.f, sj = 0.f;
  for (int d = 0; d < DH; ++d) {
    float v = (float)row[d];
    si += v * a_src[h * DH + d];
    sj += v * a_dst[h * DH + d];
  }
  e_i[idx] = si;
  e_j[idx] = sj;
}

// ---------------------------------------------------------------------------
// K5: per (b,head) max over valid j of e_j (leaky-relu is monotone, so
// rowmax_i = lrelu(e_i + Mh)).  32 threads total.
// ---------------------------------------------------------------------------
__global__ void colmax(const float* __restrict__ e_j,
                       const float* __restrict__ mask,
                       float* __restrict__ Mh) {
  int idx = threadIdx.x;           // b*HEADS + h
  if (idx >= BATCH * HEADS) return;
  int h = idx & (HEADS - 1), b = idx >> 2;
  float m = -1e30f;
  for (int j = 0; j < NB; ++j)
    if (mask[b * NB + j] != 0.f) {
      float v = e_j[(b * NB + j) * HEADS + h];
      m = v > m ? v : m;
    }
  Mh[idx] = m;
}

// ---------------------------------------------------------------------------
// K6: per (b,i,head) softmax rowmax + 1/denominator.
// Invalid i rows: reference gives uniform softmax 1/NB over ALL j.
// ---------------------------------------------------------------------------
__global__ void softmax_stats(const float* __restrict__ e_i,
                              const float* __restrict__ e_j,
                              const float* __restrict__ mask,
                              const float* __restrict__ Mh,
                              float* __restrict__ rowmax,
                              float* __restrict__ idn, int total) {
  int idx = blockIdx.x * blockDim.x + threadIdx.x;
  if (idx >= total) return;
  int h  = idx & (HEADS - 1);
  int bn = idx >> 2;
  int b  = bn >> 10;
  if (mask[bn] == 0.f) {           // invalid row -> uniform attention
    rowmax[idx] = 0.f;
    idn[idx]    = 1.f / (float)NB;
    return;
  }
  float ei  = e_i[idx];
  float rmx = ei + Mh[b * HEADS + h];
  rmx = rmx > 0.f ? rmx : 0.2f * rmx;
  float s = 0.f;
  int base = b * NB;
  for (int j = 0; j < NB; ++j) {
    if (mask[base + j] != 0.f) {
      float e = ei + e_j[(base + j) * HEADS + h];
      e = e > 0.f ? e : 0.2f * e;
      s += __expf(e - rmx);
    }
  }
  rowmax[idx] = rmx;
  idn[idx]    = 1.f / s;
}

// ---------------------------------------------------------------------------
// K7: WMMA attention aggregation.  wave = (b, head, itile of 16 rows).
// Builds the 16x32 probability A-tile in registers each j-chunk (exp on the
// fly, 1/denom folded in) and accumulates P x hT into 16x64 f32.
// hT[b,h,d,j] rows are contiguous in j => B-operand loads match A pattern.
// ---------------------------------------------------------------------------
__global__ void gat_aggregate(const _Float16* __restrict__ hT,
                              const float* __restrict__ e_i,
                              const float* __restrict__ e_j,
                              const float* __restrict__ mask,
                              const float* __restrict__ rowmax,
                              const float* __restrict__ idn,
                              float* __restrict__ hp) {
  int wave = blockIdx.x * (blockDim.x >> 5) + (threadIdx.x >> 5);
  int lane = threadIdx.x & 31;
  int itile = wave & 63;                 // NB/16 = 64 tiles
  int h     = (wave >> 6) & (HEADS - 1);
  int b     = wave >> 8;
  if (b >= BATCH) return;
  int l15  = lane & 15;
  int koff = (lane >> 4) * 8;
  int irow = itile * 16 + l15;
  int bn0  = b * NB;
  int sidx = (bn0 + irow) * HEADS + h;
  float ei_v = e_i[sidx];
  float rmx  = rowmax[sidx];
  float inv  = idn[sidx];
  bool validi = (mask[bn0 + irow] != 0.f);
  const _Float16* hTb = hT + ((size_t)(b * HEADS + h) * DH) * NB;
  v8f acc[4] = {};
  for (int j0 = 0; j0 < NB; j0 += 32) {
    v16h a;
#pragma unroll
    for (int t = 0; t < 16; ++t) {
      int j = j0 + koff + (t & 7) + ((t >> 3) << 4);
      float p;
      if (!validi) {
        p = inv;                                   // uniform 1/NB, no j-mask
      } else if (mask[bn0 + j] == 0.f) {
        p = 0.f;
      } else {
        float e = ei_v + e_j[(bn0 + j) * HEADS + h];
        e = e > 0.f ? e : 0.2f * e;
        p = __expf(e - rmx) * inv;
      }
      a[t] = (_Float16)p;
    }
#pragma unroll
    for (int nt = 0; nt < 4; ++nt) {
      const _Float16* brow = hTb + (size_t)(nt * 16 + l15) * NB;
      v8h blo = *(const v8h*)(brow + j0 + koff);
      v8h bhi = *(const v8h*)(brow + j0 + 16 + koff);
      v16h bv;
#pragma unroll
      for (int q = 0; q < 8; ++q) { bv[q] = blo[q]; bv[8 + q] = bhi[q]; }
      acc[nt] = __builtin_amdgcn_wmma_f32_16x16x32_f16(false, a, false, bv,
                                                       (short)0, acc[nt],
                                                       false, false);
    }
  }
  int mbase = (lane >> 4) * 8;
#pragma unroll
  for (int nt = 0; nt < 4; ++nt) {
#pragma unroll
    for (int v = 0; v < 8; ++v) {
      int ig = itile * 16 + mbase + v;
      int d  = nt * 16 + l15;
      hp[((size_t)(bn0 + ig)) * HID + h * DH + d] = acc[nt][v];
    }
  }
}

// ---------------------------------------------------------------------------
// K8: layer1 epilogue: residual + layernorm(256) + relu -> f16.
// ---------------------------------------------------------------------------
__global__ void ln_res_relu(const float* __restrict__ hp,
                            const float* __restrict__ res,
                            const float* __restrict__ g,
                            const float* __restrict__ be,
                            _Float16* __restrict__ out) {
  int row = blockIdx.x, t = threadIdx.x;   // 256 threads
  __shared__ float red[HID];
  float x = hp[(size_t)row * HID + t] + res[(size_t)row * HID + t];
  red[t] = x; __syncthreads();
  for (int s = 128; s > 0; s >>= 1) { if (t < s) red[t] += red[t + s]; __syncthreads(); }
  float mean = red[0] / (float)HID; __syncthreads();
  float d0 = x - mean;
  red[t] = d0 * d0; __syncthreads();
  for (int s = 128; s > 0; s >>= 1) { if (t < s) red[t] += red[t + s]; __syncthreads(); }
  float var = red[0] / (float)HID;
  float y = d0 * rsqrtf(var + 1e-5f) * g[t] + be[t];
  y = y > 0.f ? y : 0.f;
  out[(size_t)row * HID + t] = (_Float16)y;
}

// ---------------------------------------------------------------------------
// K9: layer2 epilogue: mean over heads + layernorm(64) + relu -> f32 + f16.
// ---------------------------------------------------------------------------
__global__ void head_mean_ln_relu(const float* __restrict__ hp,
                                  const float* __restrict__ g,
                                  const float* __restrict__ be,
                                  float* __restrict__ h2f,
                                  _Float16* __restrict__ h2h) {
  int row = blockIdx.x, t = threadIdx.x;   // 64 threads
  __shared__ float red[DH];
  const float* r = hp + (size_t)row * HID;
  float x = 0.25f * (r[t] + r[DH + t] + r[2 * DH + t] + r[3 * DH + t]);
  red[t] = x; __syncthreads();
  for (int s = 32; s > 0; s >>= 1) { if (t < s) red[t] += red[t + s]; __syncthreads(); }
  float mean = red[0] / (float)DH; __syncthreads();
  float d0 = x - mean;
  red[t] = d0 * d0; __syncthreads();
  for (int s = 32; s > 0; s >>= 1) { if (t < s) red[t] += red[t + s]; __syncthreads(); }
  float var = red[0] / (float)DH;
  float y = d0 * rsqrtf(var + 1e-5f) * g[t] + be[t];
  y = y > 0.f ? y : 0.f;
  h2f[(size_t)row * DH + t] = y;
  h2h[(size_t)row * DH + t] = (_Float16)y;
}

// ---------------------------------------------------------------------------
// K10: my_emb = relu(h2[b, first n with id==my_id] @ Wm + bm).
// ---------------------------------------------------------------------------
__global__ void my_emb_kernel(const float* __restrict__ agents,
                              const int* __restrict__ my_id,
                              const float* __restrict__ h2f,
                              const float* __restrict__ Wm,
                              const float* __restrict__ bm,
                              float* __restrict__ out) {
  int b = blockIdx.x, t = threadIdx.x;   // 256 threads
  __shared__ int sidx;
  if (t == 0) sidx = NB;
  __syncthreads();
  float mid = (float)my_id[b];
  for (int n = t; n < NB; n += 256)
    if (agents[((size_t)(b * NB + n)) * ED] == mid) atomicMin(&sidx, n);
  __syncthreads();
  int idx = (sidx == NB) ? 0 : sidx;
  const float* hrow = h2f + (size_t)(b * NB + idx) * DH;
  float acc = bm[t];
  for (int d = 0; d < DH; ++d) acc += hrow[d] * Wm[d * HID + t];
  acc = acc > 0.f ? acc : 0.f;
  out[OUT_EMB_OFF + (size_t)b * HID + t] = acc;
}

// ---------------------------------------------------------------------------
extern "C" void kernel_launch(void* const* d_in, const int* in_sizes, int n_in,
                              void* d_out, int out_size, void* d_ws, size_t ws_size,
                              hipStream_t stream) {
  const float* agents  = (const float*)d_in[0];
  const int*   my_id   = (const int*)d_in[1];
  const float* Wp      = (const float*)d_in[2];
  const float* bp      = (const float*)d_in[3];
  const float* W1      = (const float*)d_in[4];
  const float* a_src1  = (const float*)d_in[5];
  const float* a_dst1  = (const float*)d_in[6];
  const float* ln1_g   = (const float*)d_in[7];
  const float* ln1_b   = (const float*)d_in[8];
  const float* W2      = (const float*)d_in[9];
  const float* a_src2  = (const float*)d_in[10];
  const float* a_dst2  = (const float*)d_in[11];
  const float* ln2_g   = (const float*)d_in[12];
  const float* ln2_b   = (const float*)d_in[13];
  const float* Wa      = (const float*)d_in[14];
  const float* ba      = (const float*)d_in[15];
  const float* Wm      = (const float*)d_in[16];
  const float* bm      = (const float*)d_in[17];
  float* out = (float*)d_out;

  // -------- workspace bump allocator (256B aligned) --------
  char* base = (char*)d_ws;
  size_t off = 0;
  auto alloc = [&](size_t bytes) -> char* {
    char* p = base + off;
    off = (off + bytes + 255) & ~(size_t)255;
    return p;
  };
  float*     projF32 = (float*)    alloc((size_t)ROWS * HID * 4);
  _Float16*  projF16 = (_Float16*) alloc((size_t)ROWS * HID * 2);
  float*     maskp   = (float*)    alloc((size_t)ROWS * 4);
  _Float16*  W1t     = (_Float16*) alloc((size_t)256 * 256 * 2);
  _Float16*  W2t     = (_Float16*) alloc((size_t)256 * 256 * 2);
  _Float16*  Wat     = (_Float16*) alloc((size_t)256 * 64 * 2);
  _Float16*  hN1     = (_Float16*) alloc((size_t)ROWS * HID * 2);
  _Float16*  hT1     = (_Float16*) alloc((size_t)ROWS * HID * 2);
  float*     e_i1    = (float*)    alloc((size_t)ROWS * HEADS * 4);
  float*     e_j1    = (float*)    alloc((size_t)ROWS * HEADS * 4);
  float*     Mh1     = (float*)    alloc((size_t)BATCH * HEADS * 4);
  float*     rmx1    = (float*)    alloc((size_t)ROWS * HEADS * 4);
  float*     idn1    = (float*)    alloc((size_t)ROWS * HEADS * 4);
  float*     hp1     = (float*)    alloc((size_t)ROWS * HID * 4);
  _Float16*  h1f16   = (_Float16*) alloc((size_t)ROWS * HID * 2);
  _Float16*  hN2     = (_Float16*) alloc((size_t)ROWS * HID * 2);
  _Float16*  hT2     = (_Float16*) alloc((size_t)ROWS * HID * 2);
  float*     e_i2    = (float*)    alloc((size_t)ROWS * HEADS * 4);
  float*     e_j2    = (float*)    alloc((size_t)ROWS * HEADS * 4);
  float*     Mh2     = (float*)    alloc((size_t)BATCH * HEADS * 4);
  float*     rmx2    = (float*)    alloc((size_t)ROWS * HEADS * 4);
  float*     idn2    = (float*)    alloc((size_t)ROWS * HEADS * 4);
  float*     hp2     = (float*)    alloc((size_t)ROWS * HID * 4);
  float*     h2f32   = (float*)    alloc((size_t)ROWS * DH * 4);
  _Float16*  h2f16   = (_Float16*) alloc((size_t)ROWS * DH * 2);
  (void)ws_size; (void)n_in; (void)in_sizes; (void)out_size;

  const int eTotal = ROWS * HEADS;                     // 32768
  const int gemmBlocks = (ROWS / 16) * (HID / 16) / 8; // 8192 waves / 8 = 1024
  const int aggBlocks  = BATCH * HEADS * (NB / 16) / 8;// 2048 waves / 8 = 256

  // weights -> f16 transposed
  conv_weights<<<(131072 + 16384 + 255) / 256, 256, 0, stream>>>(W1, W2, Wa, W1t, W2t, Wat);
  // features + input projection
  prep_proj<<<ROWS, HID, 0, stream>>>(agents, Wp, bp, projF32, projF16, maskp);

  // ---------------- GAT layer 1 ----------------
  wmma_gemm16<<<gemmBlocks, 256, 0, stream>>>(projF16, W1t, nullptr,
                                              nullptr, hN1, hT1, ROWS, HID, HID, 1);
  compute_e<<<(eTotal + 255) / 256, 256, 0, stream>>>(hN1, a_src1, a_dst1, e_i1, e_j1, eTotal);
  colmax<<<1, 32, 0, stream>>>(e_j1, maskp, Mh1);
  softmax_stats<<<(eTotal + 255) / 256, 256, 0, stream>>>(e_i1, e_j1, maskp, Mh1, rmx1, idn1, eTotal);
  gat_aggregate<<<aggBlocks, 256, 0, stream>>>(hT1, e_i1, e_j1, maskp, rmx1, idn1, hp1);
  ln_res_relu<<<ROWS, HID, 0, stream>>>(hp1, projF32, ln1_g, ln1_b, h1f16);

  // ---------------- GAT layer 2 ----------------
  wmma_gemm16<<<gemmBlocks, 256, 0, stream>>>(h1f16, W2t, nullptr,
                                              nullptr, hN2, hT2, ROWS, HID, HID, 1);
  compute_e<<<(eTotal + 255) / 256, 256, 0, stream>>>(hN2, a_src2, a_dst2, e_i2, e_j2, eTotal);
  colmax<<<1, 32, 0, stream>>>(e_j2, maskp, Mh2);
  softmax_stats<<<(eTotal + 255) / 256, 256, 0, stream>>>(e_i2, e_j2, maskp, Mh2, rmx2, idn2, eTotal);
  gat_aggregate<<<aggBlocks, 256, 0, stream>>>(hT2, e_i2, e_j2, maskp, rmx2, idn2, hp2);
  head_mean_ln_relu<<<ROWS, DH, 0, stream>>>(hp2, ln2_g, ln2_b, h2f32, h2f16);

  // ---------------- outputs ----------------
  // agent_embeddings = h2 @ Wa + ba  (M=8192, K=64, N=256) straight to d_out
  wmma_gemm16<<<gemmBlocks, 256, 0, stream>>>(h2f16, Wat, ba,
                                              out, nullptr, nullptr, ROWS, HID, DH, 0);
  my_emb_kernel<<<BATCH, HID, 0, stream>>>(agents, my_id, h2f32, Wm, bm, out);
}
